// LLM_16887811408363
// MI455X (gfx1250) — compile-verified, matrix operand order checked by reading, not run on hardware
//
#include <hip/hip_runtime.h>

// ---------------------------------------------------------------------------
// MHLA GPT forward for MI455X (gfx1250): all GEMMs via v_wmma_f32_16x16x32_bf16
// (bf16 inputs, fp32 accumulate), fp32 elementwise/softmax/LN/loss.
// ---------------------------------------------------------------------------

typedef __bf16 bf16_t;
typedef __attribute__((ext_vector_type(16))) __bf16      v16bf;
typedef __attribute__((ext_vector_type(8)))  float       v8f;
typedef __attribute__((ext_vector_type(4)))  unsigned int u32x4;

#define DEV __device__ __forceinline__

namespace {
constexpr int kV = 32000, kC = 1024, kC4 = 4096, kNH = 16, kHS = 64;
constexpr int kNL = 256, kL = 8, kB = 2, kT = 1024, kM = 2048;
}

union FragU { u32x4 u[2]; v16bf v; };
union AccU  { v8f v; float f[8]; };

DEV unsigned short f2bf(float f) {
  union { float f; unsigned int u; } x; x.f = f;
  unsigned int u = x.u;
  unsigned int r = u + 0x7FFFu + ((u >> 16) & 1u);   // round to nearest even
  return (unsigned short)(r >> 16);
}

DEV float blk_sum(float v, float* sb) {
  int t = threadIdx.x;
  sb[t] = v; __syncthreads();
  for (int s = 128; s > 0; s >>= 1) { if (t < s) sb[t] += sb[t + s]; __syncthreads(); }
  float r = sb[0]; __syncthreads(); return r;
}

DEV float blk_max(float v, float* sb) {
  int t = threadIdx.x;
  sb[t] = v; __syncthreads();
  for (int s = 128; s > 0; s >>= 1) { if (t < s) sb[t] = fmaxf(sb[t], sb[t + s]); __syncthreads(); }
  float r = sb[0]; __syncthreads(); return r;
}

// ---------------------------------------------------------------------------
// Strided-batched GEMM: D[M,N] = (Csrc or 0) + bias + A[M,K] * B[N,K]^T
// A,B row-major bf16; D,Csrc row-major f32 (Csrc shares D's strides/ld).
// Block = 256 threads = 8 waves, block tile 64x64, each wave: 16x32 (2 WMMAs).
// Requires M%64==0, N%64==0, K%32==0 (always true here).
// ---------------------------------------------------------------------------
__global__ __launch_bounds__(256) void mhla_gemm(
    float* __restrict__ D, const float* __restrict__ Csrc,
    const bf16_t* __restrict__ A, const bf16_t* __restrict__ Bm,
    const float* __restrict__ bias,
    int Kdim, int lda, int ldb, int ldd,
    long long sA1, long long sA2, long long sB1, long long sB2,
    long long sD1, long long sD2, int batch2)
{
  const int z  = blockIdx.z;
  const int b1 = z / batch2;
  const int b2 = z - b1 * batch2;
  A  += (long long)b1 * sA1 + (long long)b2 * sA2;
  Bm += (long long)b1 * sB1 + (long long)b2 * sB2;
  const long long dOff = (long long)b1 * sD1 + (long long)b2 * sD2;
  D += dOff;
  if (Csrc) Csrc += dOff;

  const int w    = threadIdx.x >> 5;
  const int lane = threadIdx.x & 31;
  const int m0 = blockIdx.y * 64 + (w & 3) * 16;
  const int n0 = blockIdx.x * 64 + (w >> 2) * 32;
  const int r15 = lane & 15;
  const int hi  = lane >> 4;
  const int cm  = m0 + hi * 8;       // C/D row base for this lane
  const int cn0 = n0 + r15;
  const int cn1 = n0 + 16 + r15;

  AccU a0, a1;
  if (Csrc) {
#pragma unroll
    for (int r = 0; r < 8; ++r) {
      a0.f[r] = Csrc[(long long)(cm + r) * ldd + cn0];
      a1.f[r] = Csrc[(long long)(cm + r) * ldd + cn1];
    }
  } else {
#pragma unroll
    for (int r = 0; r < 8; ++r) { a0.f[r] = 0.f; a1.f[r] = 0.f; }
  }
  if (bias) {
    const float bv0 = bias[cn0], bv1 = bias[cn1];
#pragma unroll
    for (int r = 0; r < 8; ++r) { a0.f[r] += bv0; a1.f[r] += bv1; }
  }

  // Fragment addressing per ISA 16-bit A/B layout:
  // lane L (L<16): row L, K in {cb..cb+7, cb+16..cb+23}, cb = (L>=16)?8:0
  const int arow  = m0 + r15;
  const int brow0 = n0 + r15;
  const int brow1 = n0 + 16 + r15;
  const int koff  = hi * 8;

  for (int k0 = 0; k0 < Kdim; k0 += 32) {
    FragU fa, fb0, fb1;
    const bf16_t* pa  = A  + (long long)arow  * lda + k0 + koff;
    const bf16_t* pb0 = Bm + (long long)brow0 * ldb + k0 + koff;
    const bf16_t* pb1 = Bm + (long long)brow1 * ldb + k0 + koff;
    fa.u[0]  = *(const u32x4*)(pa);
    fa.u[1]  = *(const u32x4*)(pa + 16);
    fb0.u[0] = *(const u32x4*)(pb0);
    fb0.u[1] = *(const u32x4*)(pb0 + 16);
    fb1.u[0] = *(const u32x4*)(pb1);
    fb1.u[1] = *(const u32x4*)(pb1 + 16);
    a0.v = __builtin_amdgcn_wmma_f32_16x16x32_bf16(false, fa.v, false, fb0.v,
                                                   (short)0, a0.v, false, false);
    a1.v = __builtin_amdgcn_wmma_f32_16x16x32_bf16(false, fa.v, false, fb1.v,
                                                   (short)0, a1.v, false, false);
  }

#pragma unroll
  for (int r = 0; r < 8; ++r) {
    D[(long long)(cm + r) * ldd + cn0] = a0.f[r];
    D[(long long)(cm + r) * ldd + cn1] = a1.f[r];
  }
}

// ---------------------------------------------------------------------------
// Elementwise / reshaping helpers
// ---------------------------------------------------------------------------
__global__ __launch_bounds__(256) void mhla_f32_to_bf16(
    const float* __restrict__ in, unsigned short* __restrict__ out, long long n) {
  long long i = (long long)blockIdx.x * 256 + threadIdx.x;
  const long long stride = (long long)gridDim.x * 256;
  for (; i < n; i += stride) out[i] = f2bf(in[i]);
}

__global__ __launch_bounds__(256) void mhla_gelu_bf16(
    const float* __restrict__ in, unsigned short* __restrict__ out, long long n) {
  long long i = (long long)blockIdx.x * 256 + threadIdx.x;
  const long long stride = (long long)gridDim.x * 256;
  for (; i < n; i += stride) {
    float x = in[i];
    float g = 0.5f * x * (1.0f + erff(x * 0.70710678118654752f));
    out[i] = f2bf(g);
  }
}

// transpose f32[R,Cols] -> bf16[Cols,R]; R,Cols multiples of 32; batched.
__global__ __launch_bounds__(256) void mhla_transpose_bf16(
    const float* __restrict__ in, unsigned short* __restrict__ out,
    int R, int Cols, long long inStride, long long outStride) {
  __shared__ float tile[32][33];
  const float* ip = in + (long long)blockIdx.z * inStride;
  unsigned short* op = out + (long long)blockIdx.z * outStride;
  const int c0 = blockIdx.x * 32, r0 = blockIdx.y * 32;
  const int tx = threadIdx.x & 31, ty = threadIdx.x >> 5;  // 32 x 8
#pragma unroll
  for (int i = 0; i < 32; i += 8)
    tile[ty + i][tx] = ip[(long long)(r0 + ty + i) * Cols + c0 + tx];
  __syncthreads();
#pragma unroll
  for (int i = 0; i < 32; i += 8)
    op[(long long)(c0 + ty + i) * R + r0 + tx] = f2bf(tile[tx][ty + i]);
}

__global__ __launch_bounds__(256) void mhla_embed(
    const int* __restrict__ idx, const float* __restrict__ wte,
    const float* __restrict__ wpe, float* __restrict__ x) {
  const int m = blockIdx.x;
  const int t = m % kT;
  const int token = idx[m];
  const float* we = wte + (long long)token * kC;
  const float* pe = wpe + (long long)t * kC;
  float* xr = x + (long long)m * kC;
  for (int c = threadIdx.x; c < kC; c += 256) xr[c] = we[c] + pe[c];
}

__global__ __launch_bounds__(256) void mhla_layernorm_bf16(
    const float* __restrict__ x, const float* __restrict__ g,
    const float* __restrict__ b, unsigned short* __restrict__ out) {
  __shared__ float sb[256];
  const int m = blockIdx.x;
  const float* row = x + (long long)m * kC;
  float s = 0.f;
  for (int c = threadIdx.x; c < kC; c += 256) s += row[c];
  const float mu = blk_sum(s, sb) * (1.0f / kC);
  float v = 0.f;
  for (int c = threadIdx.x; c < kC; c += 256) { float d = row[c] - mu; v += d * d; }
  const float var = blk_sum(v, sb) * (1.0f / kC);
  const float inv = rsqrtf(var + 1e-5f);
  unsigned short* orow = out + (long long)m * kC;
  for (int c = threadIdx.x; c < kC; c += 256)
    orow[c] = f2bf((row[c] - mu) * inv * g[c] + b[c]);
}

// causal-masked scaled softmax over rows of length T; writes bf16 probs.
__global__ __launch_bounds__(256) void mhla_softmax_bf16(
    const float* __restrict__ S, unsigned short* __restrict__ P) {
  __shared__ float sb[256];
  const long long row = blockIdx.x;              // over B*NH*T
  const int qpos = (int)(row % kT);
  const float* sr = S + row * kT;
  unsigned short* pr = P + row * kT;
  float vals[4];
  float lmax = -3.0e38f;
#pragma unroll
  for (int i = 0; i < 4; ++i) {
    const int k = threadIdx.x + i * 256;
    const float v = (k <= qpos) ? sr[k] * 0.125f : -3.0e38f;
    vals[i] = v; lmax = fmaxf(lmax, v);
  }
  const float mx = blk_max(lmax, sb);
  float ls = 0.f;
#pragma unroll
  for (int i = 0; i < 4; ++i) {
    const float e = (vals[i] > -1.0e38f) ? expf(vals[i] - mx) : 0.f;
    vals[i] = e; ls += e;
  }
  const float inv = 1.0f / blk_sum(ls, sb);
#pragma unroll
  for (int i = 0; i < 4; ++i)
    pr[threadIdx.x + i * 256] = f2bf(vals[i] * inv);
}

__global__ __launch_bounds__(256) void mhla_loss_rows(
    const float* __restrict__ logits, const int* __restrict__ targets,
    float* __restrict__ rowloss) {
  __shared__ float sb[256];
  const int m = blockIdx.x;
  const float* lr = logits + (long long)m * kV;
  float lmax = -3.0e38f;
  for (int c = threadIdx.x; c < kV; c += 256) lmax = fmaxf(lmax, lr[c]);
  const float mx = blk_max(lmax, sb);
  float s = 0.f;
  for (int c = threadIdx.x; c < kV; c += 256) s += expf(lr[c] - mx);
  const float sum = blk_sum(s, sb);
  if (threadIdx.x == 0)
    rowloss[m] = mx + logf(sum) - lr[targets[m]];
}

__global__ __launch_bounds__(256) void mhla_loss_final(
    const float* __restrict__ rowloss, float* __restrict__ out) {
  __shared__ float sb[256];
  float s = 0.f;
  for (int i = threadIdx.x; i < kM; i += 256) s += rowloss[i];
  const float sum = blk_sum(s, sb);
  if (threadIdx.x == 0) out[0] = sum / (float)kM;
}

// ---------------------------------------------------------------------------
// Host orchestration
// ---------------------------------------------------------------------------
extern "C" void kernel_launch(void* const* d_in, const int* in_sizes, int n_in,
                              void* d_out, int out_size, void* d_ws, size_t ws_size,
                              hipStream_t stream) {
  (void)in_sizes; (void)n_in; (void)out_size; (void)ws_size;

  const int*   idx     = (const int*)d_in[0];
  const int*   targets = (const int*)d_in[1];
  const float* wte   = (const float*)d_in[2];
  const float* wpe   = (const float*)d_in[3];
  const float* ln1_g = (const float*)d_in[4];
  const float* ln1_b = (const float*)d_in[5];
  const float* Wq    = (const float*)d_in[6];
  const float* Wdkv  = (const float*)d_in[7];
  const float* Wuk   = (const float*)d_in[8];
  const float* Wuv   = (const float*)d_in[9];
  const float* Wo    = (const float*)d_in[10];
  const float* ln2_g = (const float*)d_in[11];
  const float* ln2_b = (const float*)d_in[12];
  const float* fc_w  = (const float*)d_in[13];
  const float* fc_b  = (const float*)d_in[14];
  const float* proj_w= (const float*)d_in[15];
  const float* proj_b= (const float*)d_in[16];
  const float* lnf_g = (const float*)d_in[17];
  const float* lnf_b = (const float*)d_in[18];
  float* out = (float*)d_out;

  // ---- workspace carve-up ----
  char* p = (char*)d_ws;
  auto alloc = [&](size_t bytes) -> void* {
    void* r = (void*)p; p += (bytes + 255) & ~(size_t)255; return r;
  };
  typedef unsigned short u16;
  float* x_f     = (float*)alloc((size_t)kM * kC * 4);
  u16*   h_bf    = (u16*)  alloc((size_t)kM * kC * 2);
  float* q_f     = (float*)alloc((size_t)kM * kC * 4);
  u16*   q_bf    = (u16*)  alloc((size_t)kM * kC * 2);
  float* ckv_f   = (float*)alloc((size_t)kM * kNL * 4);
  u16*   ckv_bf  = (u16*)  alloc((size_t)kM * kNL * 2);
  u16*   ckvT_bf = (u16*)  alloc((size_t)kM * kNL * 2);
  float* ql_f    = (float*)alloc((size_t)kM * kNH * kNL * 4);  // aliased: ctx_f
  u16*   ql_bf   = (u16*)  alloc((size_t)kM * kNH * kNL * 2);  // aliased: ctx_bf
  float* attn_f  = (float*)alloc((size_t)kB * kNH * kT * kT * 4); // aliased: mlp1_f
  u16*   attn_bf = (u16*)  alloc((size_t)kB * kNH * kT * kT * 2); // aliased: gelu_bf
  u16*   wq_bf   = (u16*)  alloc((size_t)kC * kC * 2);
  u16*   wqT_bf  = (u16*)  alloc((size_t)kC * kC * 2);
  u16*   wdkv_bf = (u16*)  alloc((size_t)kNL * kC * 2);
  u16*   wukT_bf = (u16*)  alloc((size_t)kNL * kC * 2);
  u16*   wuvT_bf = (u16*)  alloc((size_t)kNL * kC * 2);
  u16*   wo_bf   = (u16*)  alloc((size_t)kC * kC * 2);
  float* keffT_f = (float*)alloc((size_t)kNL * kC * 4);
  u16*   keffT_bf= (u16*)  alloc((size_t)kNL * kC * 2);
  float* veffT_f = (float*)alloc((size_t)kC * kNL * 4);
  u16*   veffT_bf= (u16*)  alloc((size_t)kC * kNL * 2);
  u16*   fc_bf   = (u16*)  alloc((size_t)kC4 * kC * 2);
  u16*   proj_bf = (u16*)  alloc((size_t)kC * kC4 * 2);
  u16*   wte_bf  = (u16*)  alloc((size_t)kV * kC * 2);
  float* rowloss = (float*)alloc((size_t)kM * 4);
  float* ctx_f   = ql_f;
  u16*   ctx_bf  = ql_bf;
  float* mlp1_f  = attn_f;
  u16*   gelu_bf = attn_bf;

  auto cvt = [&](const float* in, u16* o, long long n) {
    int blocks = (int)((n + 256 * 8 - 1) / (256 * 8));
    if (blocks > 8192) blocks = 8192;
    if (blocks < 1) blocks = 1;
    mhla_f32_to_bf16<<<blocks, 256, 0, stream>>>(in, o, n);
  };
  auto tra = [&](const float* in, u16* o, int R, int Cols, int batch,
                 long long inStride, long long outStride) {
    dim3 g(Cols / 32, R / 32, batch);
    mhla_transpose_bf16<<<g, 256, 0, stream>>>(in, o, R, Cols, inStride, outStride);
  };
  auto gemm = [&](float* D, const float* Csrc, const void* A, const void* Bmat,
                  const float* bias, int M, int N, int K,
                  int lda, int ldb, int ldd,
                  long long sA1, long long sA2, long long sB1, long long sB2,
                  long long sD1, long long sD2, int batch2, int batchZ) {
    dim3 g(N / 64, M / 64, batchZ);
    mhla_gemm<<<g, 256, 0, stream>>>(D, Csrc, (const bf16_t*)A, (const bf16_t*)Bmat,
                                     bias, K, lda, ldb, ldd,
                                     sA1, sA2, sB1, sB2, sD1, sD2, batch2);
  };

  // ---- one-time prep ----
  cvt(wte, wte_bf, (long long)kV * kC);
  mhla_embed<<<kM, 256, 0, stream>>>(idx, wte, wpe, x_f);

  // ---- transformer layers ----
  for (int l = 0; l < kL; ++l) {
    const float* Wq_l   = Wq    + (size_t)l * kC * kC;
    const float* Wdkv_l = Wdkv  + (size_t)l * kNL * kC;
    const float* Wuk_l  = Wuk   + (size_t)l * kC * kNL;
    const float* Wuv_l  = Wuv   + (size_t)l * kC * kNL;
    const float* Wo_l   = Wo    + (size_t)l * kC * kC;
    const float* fc_w_l = fc_w  + (size_t)l * kC4 * kC;
    const float* fc_b_l = fc_b  + (size_t)l * kC4;
    const float* pj_w_l = proj_w+ (size_t)l * kC * kC4;
    const float* pj_b_l = proj_b+ (size_t)l * kC;

    // weight prep: bf16 copies + transposes
    cvt(Wq_l,   wq_bf,   (long long)kC * kC);
    cvt(Wdkv_l, wdkv_bf, (long long)kNL * kC);
    cvt(Wo_l,   wo_bf,   (long long)kC * kC);
    cvt(fc_w_l, fc_bf,   (long long)kC4 * kC);
    cvt(pj_w_l, proj_bf, (long long)kC * kC4);
    tra(Wq_l,  wqT_bf,  kC, kC, 1, 0, 0);     // WqT[i,c]  = Wq[c,i]
    tra(Wuk_l, wukT_bf, kC, kNL, 1, 0, 0);    // WukT[l,c] = Wuk[c,l]
    tra(Wuv_l, wuvT_bf, kC, kNL, 1, 0, 0);    // WuvT[l,c] = Wuv[c,l]

    // absorbed matrices:
    // keffT[l, i] = sum_c Wuk[c,l] * Wq[c,i]      (== k_eff[h,d,l], i=h*64+d)
    gemm(keffT_f, nullptr, wukT_bf, wqT_bf, nullptr,
         kNL, kC, kC, kC, kC, kC, 0, 0, 0, 0, 0, 0, 1, 1);
    cvt(keffT_f, keffT_bf, (long long)kNL * kC);
    // veffT[i, l] = sum_c Wo[i,c] * Wuv[c,l]      (== v_eff[h,l,d], i=h*64+d)
    gemm(veffT_f, nullptr, wo_bf, wuvT_bf, nullptr,
         kC, kNL, kC, kC, kC, kNL, 0, 0, 0, 0, 0, 0, 1, 1);
    cvt(veffT_f, veffT_bf, (long long)kC * kNL);

    // h = LN1(x)
    mhla_layernorm_bf16<<<kM, 256, 0, stream>>>(x_f, ln1_g + (size_t)l * kC,
                                                ln1_b + (size_t)l * kC, h_bf);
    // q = h @ Wq^T
    gemm(q_f, nullptr, h_bf, wq_bf, nullptr,
         kM, kC, kC, kC, kC, kC, 0, 0, 0, 0, 0, 0, 1, 1);
    cvt(q_f, q_bf, (long long)kM * kC);
    // c_kv = h @ Wdkv^T
    gemm(ckv_f, nullptr, h_bf, wdkv_bf, nullptr,
         kM, kNL, kC, kC, kC, kNL, 0, 0, 0, 0, 0, 0, 1, 1);
    cvt(ckv_f, ckv_bf, (long long)kM * kNL);
    tra(ckv_f, ckvT_bf, kT, kNL, kB, (long long)kT * kNL, (long long)kNL * kT);

    // ql[m,h,l] = q_h[m,:] . keffT_h[l,:]    (batch over heads, K=HS=64)
    gemm(ql_f, nullptr, q_bf, keffT_bf, nullptr,
         kM, kNL, kHS, kC, kC, kNH * kNL,
         0, kHS, 0, kHS, 0, kNL, kNH, kNH);
    cvt(ql_f, ql_bf, (long long)kM * kNH * kNL);

    // attn logits S[b,h,q,k] = ql[b,q,h,:] . c_kv[b,k,:]   (batch B*NH, K=NL)
    gemm(attn_f, nullptr, ql_bf, ckv_bf, nullptr,
         kT, kT, kNL, kNH * kNL, kNL, kT,
         (long long)kT * kNH * kNL, kNL,
         (long long)kT * kNL, 0,
         (long long)kNH * kT * kT, (long long)kT * kT, kNH, kB * kNH);
    // causal softmax (with 1/sqrt(HS)=0.125 scale), bf16 probs
    mhla_softmax_bf16<<<kB * kNH * kT, 256, 0, stream>>>(attn_f, attn_bf);

    // ctx[b,q,h,l] = sum_k P[b,h,q,k] * c_kv[b,k,l]  (batch B*NH, K=T)
    gemm(ctx_f, nullptr, attn_bf, ckvT_bf, nullptr,
         kT, kNL, kT, kT, kT, kNH * kNL,
         (long long)kNH * kT * kT, (long long)kT * kT,
         (long long)kNL * kT, 0,
         (long long)kT * kNH * kNL, kNL, kNH, kB * kNH);
    cvt(ctx_f, ctx_bf, (long long)kM * kNH * kNL);

    // x += ctx_h @ veffT_h^T   (batch over heads, N=HS, residual via Csrc=x)
    gemm(x_f, x_f, ctx_bf, veffT_bf, nullptr,
         kM, kHS, kNL, kNH * kNL, kNL, kC,
         0, kNL, 0, (long long)kHS * kNL, 0, kHS, kNH, kNH);

    // MLP
    mhla_layernorm_bf16<<<kM, 256, 0, stream>>>(x_f, ln2_g + (size_t)l * kC,
                                                ln2_b + (size_t)l * kC, h_bf);
    gemm(mlp1_f, nullptr, h_bf, fc_bf, fc_b_l,
         kM, kC4, kC, kC, kC, kC4, 0, 0, 0, 0, 0, 0, 1, 1);
    {
      long long n = (long long)kM * kC4;
      mhla_gelu_bf16<<<8192, 256, 0, stream>>>(mlp1_f, gelu_bf, n);
    }
    gemm(x_f, x_f, gelu_bf, proj_bf, pj_b_l,
         kM, kC, kC4, kC4, kC4, kC, 0, 0, 0, 0, 0, 0, 1, 1);
  }

  // ---- final LN + tied lm_head + loss ----
  mhla_layernorm_bf16<<<kM, 256, 0, stream>>>(x_f, lnf_g, lnf_b, h_bf);
  gemm(out, nullptr, h_bf, wte_bf, nullptr,
       kM, kV, kC, kC, kC, kV, 0, 0, 0, 0, 0, 0, 1, 1);
  mhla_loss_rows<<<kM, 256, 0, stream>>>(out, targets, rowloss);
  mhla_loss_final<<<1, 256, 0, stream>>>(rowloss, out + (long long)kM * kV);
}
